// Model_39676907881556
// MI455X (gfx1250) — matrix operand order repaired and checked by reading, not compiled, not run on hardware
//
#include <hip/hip_runtime.h>
#include <hip/hip_bf16.h>

// Shapes from the reference
#define DIM 1024
#define SQN 2048          // SQ
#define BH  16            // B*H
#define QROWS (BH * SQN)  // 32768
#define KROWS (BH * DIM)  // 16384

typedef __attribute__((ext_vector_type(16))) __bf16 v16bf;
typedef __attribute__((ext_vector_type(8)))  float  v8f;
typedef __attribute__((ext_vector_type(8)))  __bf16 bf16x8;
typedef __attribute__((ext_vector_type(4)))  float  f32x4;

// ---------------------------------------------------------------------------
// Fragment loaders for gfx1250 WMMA; this code uses contiguous k order, the same for A and B
// A (16x32, 16-bit): lane<16 -> row=lane, elems = K{0..15}
//                    lane>=16 -> row=lane-16, elems = K{16..31}
// B (32x16, 16-bit): lane<16 -> col=lane, elems = K{0..15}
//                    lane>=16 -> col=lane-16, elems = K{16..31}
// C/D (16x16 f32):   VGPR j: (row j, col lane) lanes<16; (row j+8, col lane-16)
// ---------------------------------------------------------------------------

__device__ __forceinline__ v16bf load_a_f32(const float* __restrict__ A, int lda,
                                            int row0, int k0, int lane) {
  const int m  = lane & 15;
  const int kb = (lane & 16) ? 16 : 0;
  const float* p = A + (size_t)(row0 + m) * lda + (k0 + kb);
  f32x4 x0 = *(const f32x4*)(p + 0);
  f32x4 x1 = *(const f32x4*)(p + 4);
  f32x4 x2 = *(const f32x4*)(p + 8);
  f32x4 x3 = *(const f32x4*)(p + 12);
  v16bf a;
#pragma unroll
  for (int i = 0; i < 4; ++i) {
    a[i]      = (__bf16)x0[i];
    a[4 + i]  = (__bf16)x1[i];
    a[8 + i]  = (__bf16)x2[i];
    a[12 + i] = (__bf16)x3[i];
  }
  return a;
}

__device__ __forceinline__ v16bf load_a_bf16(const __bf16* __restrict__ A, int lda,
                                             int row0, int k0, int lane) {
  const int m  = lane & 15;
  const int kb = (lane & 16) ? 16 : 0;
  const __bf16* p = A + (size_t)(row0 + m) * lda + (k0 + kb);
  bf16x8 lo = *(const bf16x8*)(p + 0);
  bf16x8 hi = *(const bf16x8*)(p + 8);
  v16bf a;
#pragma unroll
  for (int i = 0; i < 8; ++i) { a[i] = lo[i]; a[8 + i] = hi[i]; }
  return a;
}

// B operand is given as T (row-major, ldt) with T[col][k] == B[k][col]
__device__ __forceinline__ v16bf load_b_f32(const float* __restrict__ T, int ldt,
                                            int col0, int k0, int lane) {
  const int c    = lane & 15;
  const int koff = (lane & 16) ? 16 : 0;
  const float* p = T + (size_t)(col0 + c) * ldt + (k0 + koff);
  f32x4 x0 = *(const f32x4*)(p + 0);
  f32x4 x1 = *(const f32x4*)(p + 4);
  f32x4 x2 = *(const f32x4*)(p + 8);
  f32x4 x3 = *(const f32x4*)(p + 12);
  v16bf b;
#pragma unroll
  for (int i = 0; i < 4; ++i) {
    b[i]      = (__bf16)x0[i];
    b[4 + i]  = (__bf16)x1[i];
    b[8 + i]  = (__bf16)x2[i];
    b[12 + i] = (__bf16)x3[i];
  }
  return b;
}

__device__ __forceinline__ v16bf load_b_bf16(const __bf16* __restrict__ T, int ldt,
                                             int col0, int k0, int lane) {
  const int c    = lane & 15;
  const int koff = (lane & 16) ? 16 : 0;
  const __bf16* p = T + (size_t)(col0 + c) * ldt + (k0 + koff);
  return *(const v16bf*)p;  // 32 contiguous bytes, 32B-aligned
}

__device__ __forceinline__ v8f wmma_bf16(v16bf a, v16bf b, v8f c) {
  // (neg_a, A, neg_b, B, c_mod, C, reuse_a, reuse_b)
  return __builtin_amdgcn_wmma_f32_16x16x32_bf16(false, a, false, b, (short)0, c,
                                                 false, false);
}

#define V8F_ZERO (v8f){0.f, 0.f, 0.f, 0.f, 0.f, 0.f, 0.f, 0.f}

// ---------------------------------------------------------------------------
// Kernel 1/2: projection  out = A(f32) x W^T + bias, output bf16
//   TRANS=false: out row-major [r][e]      (q path)
//   TRANS=true : out kbT [bh][e][n]        (k path, transposed for score B-op)
// 2x4 register blocking per wave (32x64), block = 8 waves -> 64x256 region.
// grid = (cols/256, rows/64)
// ---------------------------------------------------------------------------
template <bool TRANS>
__global__ __launch_bounds__(256) void proj_kernel(const float* __restrict__ A,
                                                   const float* __restrict__ W,
                                                   const float* __restrict__ bias,
                                                   __bf16* __restrict__ out) {
  const int lane = threadIdx.x & 31;
  const int w    = threadIdx.x >> 5;
  const int row0 = blockIdx.y * 64 + (w >> 2) * 32;
  const int col0 = blockIdx.x * 256 + (w & 3) * 64;

  v8f acc[2][4];
#pragma unroll
  for (int r = 0; r < 2; ++r)
#pragma unroll
    for (int t = 0; t < 4; ++t) acc[r][t] = V8F_ZERO;

#pragma unroll 2
  for (int k0 = 0; k0 < DIM; k0 += 32) {
    v16bf a0 = load_a_f32(A, DIM, row0, k0, lane);
    v16bf a1 = load_a_f32(A, DIM, row0 + 16, k0, lane);
#pragma unroll
    for (int t = 0; t < 4; ++t) {
      v16bf b = load_b_f32(W, DIM, col0 + 16 * t, k0, lane);  // B = W^T
      acc[0][t] = wmma_bf16(a0, b, acc[0][t]);
      acc[1][t] = wmma_bf16(a1, b, acc[1][t]);
    }
  }

#pragma unroll
  for (int r = 0; r < 2; ++r) {
#pragma unroll
    for (int t = 0; t < 4; ++t) {
      const int rr = row0 + 16 * r;
      const int cc = col0 + 16 * t;
      const float bv = bias[cc + (lane & 15)];
      if (TRANS) {
        const int bh = rr >> 10;
        const int n0 = (rr & 1023) + ((lane & 16) ? 8 : 0);
        const int e  = cc + (lane & 15);
        __bf16* p = out + ((size_t)bh << 20) + (size_t)e * DIM + n0;
        bf16x8 pk;
#pragma unroll
        for (int j = 0; j < 8; ++j) pk[j] = (__bf16)(acc[r][t][j] + bv);
        *(bf16x8*)p = pk;  // 8 contiguous bf16 (16B)
      } else {
        const int r0 = rr + ((lane & 16) ? 8 : 0);
        const int c  = cc + (lane & 15);
#pragma unroll
        for (int j = 0; j < 8; ++j)
          out[(size_t)(r0 + j) * DIM + c] = (__bf16)(acc[r][t][j] + bv);
      }
    }
  }
}

// ---------------------------------------------------------------------------
// Kernel 3: S = qb x kproj, scale, row-softmax, store P (bf16)
// One block = 16 query rows of one (b,h); 8 waves each own 8 column tiles.
// grid.x = BH * (SQN/16) = 2048
// ---------------------------------------------------------------------------
__global__ __launch_bounds__(256) void attn_score_kernel(
    const __bf16* __restrict__ qb, const __bf16* __restrict__ kbT,
    const float* __restrict__ inv_scale, __bf16* __restrict__ P) {
  const int lane = threadIdx.x & 31;
  const int w    = threadIdx.x >> 5;
  const int bh   = blockIdx.x >> 7;
  const int s0   = (blockIdx.x & 127) * 16;

  const __bf16* qrow = qb + (size_t)bh * SQN * DIM;
  const __bf16* kt   = kbT + ((size_t)bh << 20);

  v8f acc[8];
#pragma unroll
  for (int t = 0; t < 8; ++t) acc[t] = V8F_ZERO;

  for (int k0 = 0; k0 < DIM; k0 += 32) {
    v16bf a = load_a_bf16(qrow, DIM, s0, k0, lane);
#pragma unroll
    for (int t = 0; t < 8; ++t) {
      v16bf b = load_b_bf16(kt, DIM, w * 16 + t * 128, k0, lane);
      acc[t] = wmma_bf16(a, b, acc[t]);
    }
  }

  const float sc = 1.0f / inv_scale[0];

  // ---- row max over all 1024 columns ----
  float rmax[8];
#pragma unroll
  for (int j = 0; j < 8; ++j) rmax[j] = -3.0e38f;
#pragma unroll
  for (int t = 0; t < 8; ++t)
#pragma unroll
    for (int j = 0; j < 8; ++j) {
      float v = acc[t][j] * sc;
      acc[t][j] = v;
      rmax[j] = fmaxf(rmax[j], v);
    }
#pragma unroll
  for (int off = 1; off < 16; off <<= 1)
#pragma unroll
    for (int j = 0; j < 8; ++j)
      rmax[j] = fmaxf(rmax[j], __shfl_xor(rmax[j], off, 32));

  __shared__ float smax[8][16];
  __shared__ float ssum[8][16];
  const int rsel = (lane & 16) ? 8 : 0;
  if ((lane & 15) == 0) {
#pragma unroll
    for (int j = 0; j < 8; ++j) smax[w][rsel + j] = rmax[j];
  }
  __syncthreads();
  float gmax[8];
#pragma unroll
  for (int j = 0; j < 8; ++j) {
    float m = smax[0][rsel + j];
#pragma unroll
    for (int w2 = 1; w2 < 8; ++w2) m = fmaxf(m, smax[w2][rsel + j]);
    gmax[j] = m;
  }

  // ---- exp + row sum ----
  float rsum[8];
#pragma unroll
  for (int j = 0; j < 8; ++j) rsum[j] = 0.0f;
#pragma unroll
  for (int t = 0; t < 8; ++t)
#pragma unroll
    for (int j = 0; j < 8; ++j) {
      float e = __expf(acc[t][j] - gmax[j]);
      acc[t][j] = e;
      rsum[j] += e;
    }
#pragma unroll
  for (int off = 1; off < 16; off <<= 1)
#pragma unroll
    for (int j = 0; j < 8; ++j) rsum[j] += __shfl_xor(rsum[j], off, 32);
  if ((lane & 15) == 0) {
#pragma unroll
    for (int j = 0; j < 8; ++j) ssum[w][rsel + j] = rsum[j];
  }
  __syncthreads();
  float ginv[8];
#pragma unroll
  for (int j = 0; j < 8; ++j) {
    float s = ssum[0][rsel + j];
#pragma unroll
    for (int w2 = 1; w2 < 8; ++w2) s += ssum[w2][rsel + j];
    ginv[j] = 1.0f / s;
  }

  // ---- store P (bf16, row-major [bh][s][n]) ----
  __bf16* Pb = P + (size_t)bh * SQN * DIM;
  const int r0 = s0 + rsel;
  const int cl = lane & 15;
#pragma unroll
  for (int t = 0; t < 8; ++t) {
    const int col = w * 16 + t * 128 + cl;
#pragma unroll
    for (int j = 0; j < 8; ++j)
      Pb[(size_t)(r0 + j) * DIM + col] = (__bf16)(acc[t][j] * ginv[j]);
  }
}

// ---------------------------------------------------------------------------
// Kernel 4: out = P x value^T  (B-op rows = value rows, contiguous)
// 2x4 register blocking per wave (32x64), block -> 64x256 region.
// grid.x = BH * (SQN/64) * (DIM/256) = 16*32*4 = 2048
// ---------------------------------------------------------------------------
__global__ __launch_bounds__(256) void attn_out_kernel(
    const __bf16* __restrict__ P, const float* __restrict__ value,
    float* __restrict__ out) {
  const int lane = threadIdx.x & 31;
  const int w    = threadIdx.x >> 5;
  int bid = blockIdx.x;
  const int colblk = bid & 3;   bid >>= 2;
  const int rowblk = bid & 31;
  const int bh     = bid >> 5;
  const int row0 = rowblk * 64 + (w >> 2) * 32;
  const int col0 = colblk * 256 + (w & 3) * 64;

  const __bf16* Pb = P + (size_t)bh * SQN * DIM;
  const float*  vb = value + ((size_t)bh << 20);

  v8f acc[2][4];
#pragma unroll
  for (int r = 0; r < 2; ++r)
#pragma unroll
    for (int t = 0; t < 4; ++t) acc[r][t] = V8F_ZERO;

#pragma unroll 2
  for (int k0 = 0; k0 < DIM; k0 += 32) {
    v16bf a0 = load_a_bf16(Pb, DIM, row0, k0, lane);
    v16bf a1 = load_a_bf16(Pb, DIM, row0 + 16, k0, lane);
#pragma unroll
    for (int t = 0; t < 4; ++t) {
      v16bf b = load_b_f32(vb, DIM, col0 + 16 * t, k0, lane);  // B = value^T
      acc[0][t] = wmma_bf16(a0, b, acc[0][t]);
      acc[1][t] = wmma_bf16(a1, b, acc[1][t]);
    }
  }

  float* ob = out + (size_t)bh * SQN * DIM;
#pragma unroll
  for (int r = 0; r < 2; ++r) {
#pragma unroll
    for (int t = 0; t < 4; ++t) {
      const int r0 = row0 + 16 * r + ((lane & 16) ? 8 : 0);
      const int c  = col0 + 16 * t + (lane & 15);
#pragma unroll
      for (int j = 0; j < 8; ++j) ob[(size_t)(r0 + j) * DIM + c] = acc[r][t][j];
    }
  }
}

// ---------------------------------------------------------------------------
extern "C" void kernel_launch(void* const* d_in, const int* in_sizes, int n_in,
                              void* d_out, int out_size, void* d_ws, size_t ws_size,
                              hipStream_t stream) {
  (void)in_sizes; (void)n_in; (void)out_size; (void)ws_size;
  const float* query     = (const float*)d_in[0];
  const float* key       = (const float*)d_in[1];
  const float* value     = (const float*)d_in[2];
  const float* Wq        = (const float*)d_in[3];
  const float* bq        = (const float*)d_in[4];
  const float* Wk        = (const float*)d_in[5];
  const float* bk        = (const float*)d_in[6];
  const float* inv_scale = (const float*)d_in[7];

  // Workspace layout (bf16): qb[32768x1024] | kbT[16x1024x1024] | P[16x2048x1024]
  __bf16* qb  = (__bf16*)d_ws;
  __bf16* kbT = qb + (size_t)QROWS * DIM;
  __bf16* Pb  = kbT + (size_t)KROWS * DIM;

  proj_kernel<false><<<dim3(DIM / 256, QROWS / 64), 256, 0, stream>>>(query, Wq, bq, qb);
  proj_kernel<true ><<<dim3(DIM / 256, KROWS / 64), 256, 0, stream>>>(key,   Wk, bk, kbT);
  attn_score_kernel<<<BH * (SQN / 16), 256, 0, stream>>>(qb, kbT, inv_scale, Pb);
  attn_out_kernel<<<BH * (SQN / 64) * (DIM / 256), 256, 0, stream>>>(Pb, value,
                                                                     (float*)d_out);
}
